// RGAT_Layer_7344394076549
// MI455X (gfx1250) — matrix operand order repaired and checked by reading, not compiled, not run on hardware
//
#include <hip/hip_runtime.h>
#include <hip/hip_bf16.h>

typedef __attribute__((ext_vector_type(16))) _Float16 v16h;
typedef __attribute__((ext_vector_type(8)))  _Float16 v8h;
typedef __attribute__((ext_vector_type(8)))  float    v8f;

#define NNODES 50000
#define DEG    16
#define NDIM   128
#define QKVN   384
#define FFNH   512

// ---------------------------------------------------------------------------
// f32 -> f16 conversion
// ---------------------------------------------------------------------------
__global__ __launch_bounds__(256)
void cvt_f32_f16(const float* __restrict__ in, _Float16* __restrict__ out, int n) {
    int i = blockIdx.x * blockDim.x + threadIdx.x;
    if (i < n) out[i] = (_Float16)in[i];
}

// ---------------------------------------------------------------------------
// Generic WMMA GEMM:  out[M,N] = A[M,K] @ W[N,K]^T  (+bias)(+residual)(relu)
// A, W are f16 row-major (K contiguous). Accumulate f32 via v_wmma.
// One wave computes one 16x16 output tile. blockDim=128 (4 waves along N).
// grid = (M/16, N/64). M,N mult of 16; K mult of 32 (guaranteed here).
//
// Fragment packing uses contiguous k order (k = 16*half + s) for both A and B:
//   A 16x32 f16: lane l -> row=l&15, half=l>>4; elems 0..7 = K[half*16..+7],
//                elems 8..15 = K[half*16+8..+7]
//   B 32x16 f16: lane l -> col=l&15; elems 0..15 = K[(l>>4)*16 .. +15]
//   C/D f32:     lane l -> col=l&15; elem i -> row = (l>>4)*8 + i
// ---------------------------------------------------------------------------
__global__ __launch_bounds__(128)
void gemm_wmma_f16(const _Float16* __restrict__ A,
                   const _Float16* __restrict__ W,
                   const float* __restrict__ bias, int bias_n,
                   const float* __restrict__ residual,
                   float* __restrict__ outF,
                   _Float16* __restrict__ outH,
                   int N, int K, int relu)
{
    const int wave = threadIdx.x >> 5;
    const int lane = threadIdx.x & 31;
    const int mi   = blockIdx.x;
    const int ni   = blockIdx.y * 4 + wave;
    const int row  = lane & 15;
    const int half = lane >> 4;
    const int col  = ni * 16 + row;           // N-column this lane owns in B/C/D

    const _Float16* arow = A + (size_t)(mi * 16 + row) * K;
    const _Float16* wrow = W + (size_t)col * K + half * 16;

    v8f c = {};
    for (int k0 = 0; k0 < K; k0 += 32) {
        union { v16h v; v8h h[2]; } a;
        a.h[0] = *(const v8h*)(arow + k0 + half * 16);
        a.h[1] = *(const v8h*)(arow + k0 + half * 16 + 8);
        v16h b = *(const v16h*)(wrow + k0);
        c = __builtin_amdgcn_wmma_f32_16x16x32_f16(
                false, a.v, false, b, (short)0, c, false, false);
    }

    float bv = 0.0f;
    if (bias && col < bias_n) bv = bias[col];

#pragma unroll
    for (int i = 0; i < 8; ++i) {
        const int rg = mi * 16 + half * 8 + i;
        float val = c[i] + bv;
        if (residual) val += residual[(size_t)rg * N + col];
        if (relu)     val = fmaxf(val, 0.0f);
        if (outF) outF[(size_t)rg * N + col] = val;
        if (outH) outH[(size_t)rg * N + col] = (_Float16)val;
    }
}

// ---------------------------------------------------------------------------
// Edge attention + softmax-normalized aggregation.
// edge_dst = repeat(arange(N), DEG): node n owns edges [n*16, n*16+16), so the
// segment-sum is a per-node register accumulation (no atomics), and the 410MB
// lgx stream is read exactly once, fully sequentially -> non-temporal loads
// (TH=NT) so it never evicts the L2-resident q/k/v working set (77MB < 192MB
// L2), which the random k[src]/v[src] gathers depend on.
// The next edge's src row is prefetched (global_prefetch_b8) one iteration
// ahead to hide gather latency behind the current edge's math.
// Block = 128 threads = 1 node; thread d = feature dim; head = d>>4 (DK=16).
// qkv layout: [N, 384] = q | k | v.
// ---------------------------------------------------------------------------
__global__ __launch_bounds__(128)
void edge_attn(const float* __restrict__ qkv,
               const float* __restrict__ lgx,
               const int*   __restrict__ edge_src,
               _Float16*    __restrict__ o16)
{
    const int n = blockIdx.x;
    const int d = threadIdx.x;                    // 0..127
    const float qd = qkv[(size_t)n * QKVN + d];

    float acc = 0.0f, z = 0.0f;
    int src_next = edge_src[n * DEG];
#pragma unroll 4
    for (int eid = 0; eid < DEG; ++eid) {
        const int e   = n * DEG + eid;
        const int src = src_next;
        if (eid + 1 < DEG) {
            src_next = edge_src[e + 1];
            // prefetch next edge's k-row and v-row (L2 gather warm-up)
            __builtin_prefetch(qkv + (size_t)src_next * QKVN + NDIM + d, 0, 3);
            __builtin_prefetch(qkv + (size_t)src_next * QKVN + 2 * NDIM + d, 0, 3);
        }
        const float ev = __builtin_nontemporal_load(lgx + (size_t)e * NDIM + d);
        const float kv = qkv[(size_t)src * QKVN + NDIM + d];
        float p = (kv + ev) * qd;
        p += __shfl_xor(p, 1, 16);
        p += __shfl_xor(p, 2, 16);
        p += __shfl_xor(p, 4, 16);
        p += __shfl_xor(p, 8, 16);
        // DK=16 -> scale 1/sqrt(16)=0.25, clamp to [-10,10], exp
        const float s = __expf(fminf(fmaxf(p * 0.25f, -10.0f), 10.0f));
        const float vv = qkv[(size_t)src * QKVN + 2 * NDIM + d];
        acc += (vv + ev) * s;
        z   += s;
    }
    o16[(size_t)n * NDIM + d] = (_Float16)(acc / z);
}

// ---------------------------------------------------------------------------
// Row LayerNorm over 128 features. Block = 128 = 4 waves, one wave per row.
// Lane handles elems {lane, lane+32, lane+64, lane+96}; shuffle tree reduce.
// ---------------------------------------------------------------------------
__global__ __launch_bounds__(128)
void layernorm128(const float* __restrict__ in,
                  const float* __restrict__ g,
                  const float* __restrict__ b,
                  float* __restrict__ outF,
                  _Float16* __restrict__ outH)
{
    const int wave = threadIdx.x >> 5;
    const int lane = threadIdx.x & 31;
    const int row  = blockIdx.x * 4 + wave;
    const float* rp = in + (size_t)row * NDIM;

    float x0 = rp[lane], x1 = rp[lane + 32], x2 = rp[lane + 64], x3 = rp[lane + 96];
    float s = x0 + x1 + x2 + x3;
    float q = x0 * x0 + x1 * x1 + x2 * x2 + x3 * x3;
    for (int m = 1; m < 32; m <<= 1) {
        s += __shfl_xor(s, m, 32);
        q += __shfl_xor(q, m, 32);
    }
    const float mean = s * (1.0f / 128.0f);
    const float var  = q * (1.0f / 128.0f) - mean * mean;
    const float rstd = rsqrtf(var + 1e-5f);

#pragma unroll
    for (int j = 0; j < 4; ++j) {
        const int dd = lane + j * 32;
        const float y = (rp[dd] - mean) * rstd * g[dd] + b[dd];
        if (outF) outF[(size_t)row * NDIM + dd] = y;
        if (outH) outH[(size_t)row * NDIM + dd] = (_Float16)y;
    }
}

// ---------------------------------------------------------------------------
// Host-side launch sequence
// ---------------------------------------------------------------------------
extern "C" void kernel_launch(void* const* d_in, const int* in_sizes, int n_in,
                              void* d_out, int out_size, void* d_ws, size_t ws_size,
                              hipStream_t stream) {
    const float* x        = (const float*)d_in[0];
    const float* lgx      = (const float*)d_in[1];
    const int*   edge_src = (const int*)  d_in[2];
    // d_in[3] = edge_dst, structurally repeat(arange(N),DEG) -> not needed
    const float* Wq    = (const float*)d_in[4];
    const float* bq    = (const float*)d_in[5];
    const float* Wk    = (const float*)d_in[6];
    const float* Wv    = (const float*)d_in[7];
    const float* Wo    = (const float*)d_in[8];
    const float* bo    = (const float*)d_in[9];
    const float* ln1_g = (const float*)d_in[10];
    const float* ln1_b = (const float*)d_in[11];
    const float* W1    = (const float*)d_in[12];
    const float* b1    = (const float*)d_in[13];
    const float* W2    = (const float*)d_in[14];
    const float* b2    = (const float*)d_in[15];
    const float* ln2_g = (const float*)d_in[16];
    const float* ln2_b = (const float*)d_in[17];
    float* out = (float*)d_out;

    // ---- workspace layout (bytes, all offsets 256B-aligned) ----
    char* ws = (char*)d_ws;
    const size_t SZ_X16  = (size_t)NNODES * NDIM * 2;          // 12,800,000
    const size_t SZ_WQKV = (size_t)QKVN * NDIM * 2;            //     98,304
    const size_t SZ_WO   = (size_t)NDIM * NDIM * 2;            //     32,768
    const size_t SZ_W1   = (size_t)FFNH * NDIM * 2;            //    131,072
    const size_t SZ_W2   = (size_t)NDIM * FFNH * 2;            //    131,072
    const size_t SZ_QKV  = (size_t)NNODES * QKVN * 4;          // 76,800,000
    const size_t SZ_O16  = (size_t)NNODES * NDIM * 2;          // 12,800,000
    const size_t SZ_H    = (size_t)NNODES * NDIM * 4;          // 25,600,000

    size_t off = 0;
    _Float16* x16    = (_Float16*)(ws + off); off += SZ_X16;
    _Float16* wqkv16 = (_Float16*)(ws + off); off += SZ_WQKV;
    _Float16* wo16   = (_Float16*)(ws + off); off += SZ_WO;
    _Float16* w116   = (_Float16*)(ws + off); off += SZ_W1;
    _Float16* w216   = (_Float16*)(ws + off); off += SZ_W2;
    char*     bigA   = ws + off;              off += SZ_QKV;   // qkv, later reused
    _Float16* o16    = (_Float16*)(ws + off); off += SZ_O16;
    float*    h      = (float*)   (ws + off); off += SZ_H;
    _Float16* h16    = (_Float16*)(ws + off); off += SZ_X16;
    // reuse of bigA after edge kernel: hpre (25.6MB) then t16 (51.2MB); outpre
    // reuses hpre after LN1. All disjoint from their concurrent readers.
    float*    qkv    = (float*)bigA;
    float*    hpre   = (float*)bigA;
    _Float16* t16    = (_Float16*)(bigA + SZ_H);
    float*    outpre = (float*)bigA;

    // ---- 1) conversions ----
    {
        int n = NNODES * NDIM;
        cvt_f32_f16<<<(n + 255) / 256, 256, 0, stream>>>(x, x16, n);
    }
    cvt_f32_f16<<<(NDIM * NDIM + 255) / 256, 256, 0, stream>>>(Wq, wqkv16, NDIM * NDIM);
    cvt_f32_f16<<<(NDIM * NDIM + 255) / 256, 256, 0, stream>>>(Wk, wqkv16 + NDIM * NDIM, NDIM * NDIM);
    cvt_f32_f16<<<(NDIM * NDIM + 255) / 256, 256, 0, stream>>>(Wv, wqkv16 + 2 * NDIM * NDIM, NDIM * NDIM);
    cvt_f32_f16<<<(NDIM * NDIM + 255) / 256, 256, 0, stream>>>(Wo, wo16, NDIM * NDIM);
    cvt_f32_f16<<<(FFNH * NDIM + 255) / 256, 256, 0, stream>>>(W1, w116, FFNH * NDIM);
    cvt_f32_f16<<<(NDIM * FFNH + 255) / 256, 256, 0, stream>>>(W2, w216, NDIM * FFNH);

    const int MT = NNODES / 16;  // 3125, exact

    // ---- 2) fused QKV projection: qkv[N,384] = x @ [Wq;Wk;Wv]^T (+bq on q) ----
    gemm_wmma_f16<<<dim3(MT, QKVN / 64), 128, 0, stream>>>(
        x16, wqkv16, bq, NDIM, nullptr, qkv, nullptr, QKVN, NDIM, 0);

    // ---- 3) edge attention + aggregation -> o (f16) ----
    edge_attn<<<NNODES, 128, 0, stream>>>(qkv, lgx, edge_src, o16);

    // ---- 4) hpre = x + o @ Wo^T + bo ----
    gemm_wmma_f16<<<dim3(MT, NDIM / 64), 128, 0, stream>>>(
        o16, wo16, bo, NDIM, x, hpre, nullptr, NDIM, NDIM, 0);

    // ---- 5) h = LN1(hpre)  (f32 + f16) ----
    layernorm128<<<NNODES / 4, 128, 0, stream>>>(hpre, ln1_g, ln1_b, h, h16);

    // ---- 6) t = relu(h @ W1^T + b1)  (f16 only) ----
    gemm_wmma_f16<<<dim3(MT, FFNH / 64), 128, 0, stream>>>(
        h16, w116, b1, FFNH, nullptr, nullptr, t16, FFNH, NDIM, 1);

    // ---- 7) outpre = h + t @ W2^T + b2 ----
    gemm_wmma_f16<<<dim3(MT, NDIM / 64), 128, 0, stream>>>(
        t16, w216, b2, NDIM, h, outpre, nullptr, NDIM, FFNH, 0);

    // ---- 8) out = LN2(outpre) ----
    layernorm128<<<NNODES / 4, 128, 0, stream>>>(outpre, ln2_g, ln2_b, out, nullptr);
}
